// graph_constructor_24008867184897
// MI455X (gfx1250) — 8-bit matrix operand order repaired and checked by reading, compiled, not run on hardware
//
#include <hip/hip_runtime.h>
#include <hip/hip_bf16.h>

typedef __attribute__((ext_vector_type(8))) int v8i;

#define B_DIM 64
#define N_DIM 1024
#define D_DIM 256

#define CORR_OFF 4161600
#define INV_NORM2 (1.0f/1398080.0f)

__global__ __launch_bounds__(256) void rank_kernel(
    const float* __restrict__ x, const float* __restrict__ conv_w,
    const float* __restrict__ conv_b, const float* __restrict__ time_g,
    const float* __restrict__ time_b, unsigned char* __restrict__ ranks)
{
    __shared__ float xv[26];
    __shared__ float sv[256];
    __shared__ float wred[16];
    const int d  = threadIdx.x;
    const int bn = blockIdx.x;
    const int b  = bn >> 10, n = bn & 1023;

    if (d < 26) {
        int c = d / 13, t = d - c * 13;
        xv[d] = x[(((size_t)b * 2 + c) * N_DIM + n) * 13 + t];
    }
    __syncthreads();

    float e = conv_b[d];
    #pragma unroll
    for (int i = 0; i < 26; ++i) e = fmaf(xv[i], conv_w[d * 26 + i], e);

    float s = e, q = e * e;
    #pragma unroll
    for (int m = 16; m >= 1; m >>= 1) {
        s += __shfl_xor(s, m, 32);
        q += __shfl_xor(q, m, 32);
    }
    const int wave = d >> 5, lane = d & 31;
    if (lane == 0) { wred[wave] = s; wred[8 + wave] = q; }
    __syncthreads();
    if (d == 0) {
        float ts = 0.f, tq = 0.f;
        #pragma unroll
        for (int i = 0; i < 8; ++i) { ts += wred[i]; tq += wred[8 + i]; }
        wred[0] = ts; wred[8] = tq;
    }
    __syncthreads();
    const float mu   = wred[0] * (1.0f / 256.0f);
    const float var  = wred[8] * (1.0f / 256.0f) - mu * mu;
    const float v    = (e - mu) * rsqrtf(var + 1e-5f) * time_g[d] + time_b[d];
    sv[d] = v;
    __syncthreads();

    int r = 0;
    for (int j = 0; j < 256; ++j) {
        float o = sv[j];
        r += (o < v) || (o == v && j < d);
    }
    ranks[(size_t)bn * D_DIM + d] = (unsigned char)r;
}

__global__ __launch_bounds__(512) void corr_ln_kernel(
    const unsigned char* __restrict__ ranks,
    const float* __restrict__ static_g, const float* __restrict__ static_b,
    float* __restrict__ out)
{
    extern __shared__ float adj[];
    const int tid  = threadIdx.x;
    const int wave = tid >> 5, lane = tid & 31;
    const int half = lane >> 4, l16 = lane & 15;
    const int b    = blockIdx.x >> 4;
    const int row0 = (blockIdx.x & 15) * 64;
    const int mt   = wave & 3;
    const int ng   = wave >> 2;
    const size_t rbase = (size_t)b * N_DIM * D_DIM;

    v8i a[4];
    {
        const unsigned char* ap = ranks + rbase + (size_t)(row0 + mt * 16 + l16) * D_DIM;
        #pragma unroll
        for (int kc = 0; kc < 4; ++kc) {
            #pragma unroll
            for (int v = 0; v < 8; ++v) {
                int k = kc * 64 + (v >> 2) * 32 + ((v >> 1) & 1) * 16 + (v & 1) * 4 + half * 8;
                a[kc][v] = *(const int*)(ap + k);
            }
        }
    }

    for (int j = 0; j < 16; ++j) {
        const int nt = ng * 16 + j;
        const unsigned char* bp = ranks + rbase + (size_t)(nt * 16 + l16) * D_DIM;
        v8i acc = (v8i){0, 0, 0, 0, 0, 0, 0, 0};
        #pragma unroll
        for (int kc = 0; kc < 4; ++kc) {
            v8i bm;
            #pragma unroll
            for (int v = 0; v < 8; ++v) {
                int k = kc * 64 + (v >> 2) * 32 + ((v >> 1) & 1) * 16 + (v & 1) * 4 + half * 8;
                bm[v] = *(const int*)(bp + k);
            }
            acc = __builtin_amdgcn_wmma_i32_16x16x64_iu8(
                 false, a[kc],  false, bm, acc,
                 false,  false);
        }
        #pragma unroll
        for (int p = 0; p < 8; ++p) {
            int   lrow = mt * 16 + half * 8 + p;
            float corr = (float)(acc[p] - CORR_OFF);
            adj[lrow * N_DIM + nt * 16 + l16] = fabsf(corr) * INV_NORM2;
        }
    }
    __syncthreads();

    #pragma unroll
    for (int k = 0; k < 4; ++k) {
        const int lrow = wave * 4 + k;
        float s = 0.f, q = 0.f;
        #pragma unroll 8
        for (int i = 0; i < 32; ++i) {
            float v = adj[lrow * N_DIM + lane + 32 * i];
            s += v; q += v * v;
        }
        #pragma unroll
        for (int m = 16; m >= 1; m >>= 1) {
            s += __shfl_xor(s, m, 32);
            q += __shfl_xor(q, m, 32);
        }
        const float mu   = s * (1.0f / 1024.0f);
        const float var  = q * (1.0f / 1024.0f) - mu * mu;
        const float rstd = rsqrtf(var + 1e-5f);
        const size_t obase = ((size_t)b * N_DIM + row0 + lrow) * N_DIM;
        #pragma unroll 8
        for (int i = 0; i < 32; ++i) {
            int   col = lane + 32 * i;
            float v = (adj[lrow * N_DIM + col] - mu) * rstd * static_g[col] + static_b[col];
            out[obase + col] = v > 0.f ? v : 0.f;
        }
    }
}

extern "C" void kernel_launch(void* const* d_in, const int* in_sizes, int n_in,
                              void* d_out, int out_size, void* d_ws, size_t ws_size,
                              hipStream_t stream) {
    const float* x        = (const float*)d_in[0];
    const float* conv_w   = (const float*)d_in[1];
    const float* conv_b   = (const float*)d_in[2];
    const float* time_g   = (const float*)d_in[3];
    const float* time_b   = (const float*)d_in[4];
    const float* static_g = (const float*)d_in[5];
    const float* static_b = (const float*)d_in[6];
    float* out = (float*)d_out;
    unsigned char* ranks = (unsigned char*)d_ws;

    rank_kernel<<<B_DIM * N_DIM, 256, 0, stream>>>(
        x, conv_w, conv_b, time_g, time_b, ranks);

    corr_ln_kernel<<<B_DIM * 16, 512, 64 * 1024 * (int)sizeof(float), stream>>>(
        ranks, static_g, static_b, out);

    (void)in_sizes; (void)n_in; (void)out_size; (void)ws_size;
}
